// biLSTM_CRF_29317446762909
// MI455X (gfx1250) — matrix operand order repaired and checked by reading, not compiled, not run on hardware
//
#include <hip/hip_runtime.h>
#include <hip/hip_bf16.h>
#include <math.h>

typedef _Float16 half_t;
typedef __attribute__((ext_vector_type(16))) _Float16 v16h;
typedef __attribute__((ext_vector_type(8)))  _Float16 v8h;
typedef __attribute__((ext_vector_type(8)))  float    v8f;
typedef __attribute__((ext_vector_type(4)))  float    v4f;

union HV { v16h v; v8h h[2]; };

// Problem constants
constexpr int   T_   = 512;
constexpr int   B_   = 128;
constexpr int   D_   = 50;
constexpr int   H_   = 200;
constexpr int   K_   = 25;
constexpr int   Dp   = 64;    // padded embedding dim (2 k-tiles of 32)
constexpr int   Hp   = 208;   // padded gate chunk (13 n-tiles of 16)
constexpr int   NT   = 52;    // 4 gates * 13 tiles
constexpr int   KTR  = 7;     // recurrent K tiles (224 = pad of 200)
constexpr int   KREC = 224;
constexpr int   IN1  = 416;   // padded layer-1 input (13 k-tiles of 32)
constexpr int   MT   = (T_*B_)/16;      // 4096 row tiles
constexpr long  XW_DIR = (long)T_*NT*8*256; // floats per direction in xw buffer

__device__ __forceinline__ float sigm(float x) { return 1.0f / (1.0f + __expf(-x)); }

// ---------------------------------------------------------------------------
// Embedding gather + f32->f16, padded to Dp columns.  xeh is (T*B, Dp) row-major.
// ---------------------------------------------------------------------------
__global__ __launch_bounds__(256) void embed_kernel(const int* __restrict__ x,
                                                    const float* __restrict__ emb,
                                                    half_t* __restrict__ xeh) {
  long idx = (long)blockIdx.x * blockDim.x + threadIdx.x;
  long total = (long)T_ * B_ * Dp;
  if (idx >= total) return;
  int col = (int)(idx & (Dp - 1));
  long row = idx >> 6;          // row = t*128 + b
  int t = (int)(row >> 7), b = (int)(row & 127);
  float v = 0.f;
  if (col < D_) {
    int tok = x[(long)b * T_ + t];
    v = emb[(long)tok * D_ + col];
  }
  xeh[idx] = (half_t)v;
}

// ---------------------------------------------------------------------------
// Pre-swizzle a weight matrix W[out, Ksrc] into WMMA B fragments:
// dst[dir][kt][nt][lane][16 halves], lane: N = nt*16 + lane%16,
// K = kt*32 + (lane/16)*8 + (j%8) + (j/8)*16   (matches 16-bit B-matrix 32x16 VGPR layout).
// N index is grouped in "chunks" (gates) of rowsPerChunk padded to tilesPerChunk*16.
// ---------------------------------------------------------------------------
__global__ __launch_bounds__(256) void prep_bfrag(const float* __restrict__ W,
                                                  half_t* __restrict__ dst,
                                                  int ktiles, int ntiles,
                                                  int rowsPerChunk, int tilesPerChunk,
                                                  int Ksrc, long total) {
  long idx = (long)blockIdx.x * blockDim.x + threadIdx.x;
  if (idx >= total) return;
  int jj   = (int)(idx & 15);
  int lane = (int)((idx >> 4) & 31);
  long f   = idx >> 9;
  int nt   = (int)(f % ntiles); f /= ntiles;
  int kt   = (int)(f % ktiles); f /= ktiles;
  int dir  = (int)f;
  int k    = kt * 32 + (lane >> 4) * 8 + (jj & 7) + (jj >> 3) * 16;
  int chunk = nt / tilesPerChunk;
  int hh    = (nt % tilesPerChunk) * 16 + (lane & 15);
  float v = 0.f;
  if (hh < rowsPerChunk && k < Ksrc) {
    long rowsTotal = (long)(ntiles / tilesPerChunk) * rowsPerChunk;
    long srcRow = (long)chunk * rowsPerChunk + hh;
    v = W[((long)dir * rowsTotal + srcRow) * Ksrc + k];
  }
  dst[idx] = (half_t)v;
}

// bias (2,2,4H) -> padded per-direction (2, 832) with bsum = b[:,0]+b[:,1]
__global__ __launch_bounds__(256) void prep_bias(const float* __restrict__ b,
                                                 float* __restrict__ dst) {
  int idx = blockIdx.x * blockDim.x + threadIdx.x;
  if (idx >= 2 * 4 * Hp) return;
  int dir = idx / (4 * Hp), p = idx % (4 * Hp);
  int gate = p / Hp, r = p % Hp;
  float v = 0.f;
  if (r < H_) {
    int o = gate * H_ + r;
    v = b[((long)dir * 2 + 0) * (4 * H_) + o] + b[((long)dir * 2 + 1) * (4 * H_) + o];
  }
  dst[idx] = v;
}

__global__ __launch_bounds__(256) void zero_half(half_t* __restrict__ p, long n) {
  for (long i = (long)blockIdx.x * blockDim.x + threadIdx.x; i < n;
       i += (long)gridDim.x * blockDim.x)
    p[i] = (half_t)0.0f;
}

__global__ void zero_out(float* o) { if (threadIdx.x == 0 && blockIdx.x == 0) *o = 0.f; }

// ---------------------------------------------------------------------------
// Input-projection GEMM:  xw[dir] = A(M,Kp) * Bfrag + bias.  One wave = one
// 16x16 C tile.  Output stored in C-fragment tile order so the recurrent
// kernel can load accumulators with two b128 loads and no shuffles:
// xw[dir][t][nt][rt][lane][8 f32].
// ---------------------------------------------------------------------------
__global__ __launch_bounds__(256) void wmma_gemm_xw(const half_t* __restrict__ A, int astride,
                                                    const half_t* __restrict__ Bf, int ktiles,
                                                    const float* __restrict__ biasPad,
                                                    float* __restrict__ Out, int dirs) {
  int wid = threadIdx.x >> 5, lane = threadIdx.x & 31;
  long gw = (long)blockIdx.x * 8 + wid;
  long total = (long)dirs * MT * NT;
  if (gw >= total) return;
  int nt = (int)(gw % NT); long r1 = gw / NT;
  int mt = (int)(r1 % MT); int dir = (int)(r1 / MT);
  int lan15 = lane & 15;
  float bv = biasPad[(long)dir * 4 * Hp + nt * 16 + lan15];
  v8f acc = {bv, bv, bv, bv, bv, bv, bv, bv};
  int khalfA = (lane >> 4) * 8;
  const half_t* Arow = A + (long)(mt * 16 + lan15) * astride;
  const half_t* Bbase = Bf + (((long)dir * ktiles) * NT + nt) * 512 + lane * 16;
  for (int kt = 0; kt < ktiles; ++kt) {
    HV a;
    a.h[0] = *(const v8h*)(Arow + kt * 32 + khalfA);
    a.h[1] = *(const v8h*)(Arow + kt * 32 + khalfA + 16);
    v16h b = *(const v16h*)(Bbase + (long)kt * NT * 512);
    acc = __builtin_amdgcn_wmma_f32_16x16x32_f16(false, a.v, false, b, (short)0, acc,
                                                 false, false);
  }
  int t = mt >> 3, rt = mt & 7;
  float* op = Out + (long)dir * XW_DIR + (((long)t * NT + nt) * 8 + rt) * 256 + lane * 8;
  v4f lo = {acc[0], acc[1], acc[2], acc[3]};
  v4f hi = {acc[4], acc[5], acc[6], acc[7]};
  *(v4f*)op = lo;
  *(v4f*)(op + 4) = hi;
}

// ---------------------------------------------------------------------------
// Persistent LSTM recurrence, one workgroup per direction (blockIdx.x = dir).
// 8 waves; wave w owns batch row-tile w (rows 16w..16w+15) for all 512 steps:
//   - h staged in LDS as next step's A matrix (128 x 224 f16, zero-padded)
//   - c kept in LDS in fragment layout (8 x 13 tiles x 256 f32)
//   - Whh read as pre-swizzled B fragments from L2
//   - per step, per h-tile: 4 gate accumulators, 7 K-tiles -> 28 WMMAs
// ---------------------------------------------------------------------------
__global__ __launch_bounds__(256) void lstm_rec(const float* __restrict__ xw,
                                                const half_t* __restrict__ WhhF,
                                                half_t* __restrict__ hOut) {
  extern __shared__ char smem[];
  half_t* hS = (half_t*)smem;                                   // [128][224]
  float*  cS = (float*)(smem + (size_t)B_ * KREC * sizeof(half_t)); // [8][13][256]
  int dir = blockIdx.x;
  int wid = threadIdx.x >> 5, lane = threadIdx.x & 31;
  // zero LDS (h=0, c=0, pads)
  {
    int* si = (int*)smem;
    int nInt = (B_ * KREC * 2 + 8 * 13 * 256 * 4) / 4;
    for (int i = threadIdx.x; i < nInt; i += 256) si[i] = 0;
  }
  __syncthreads();

  const float*  xwD = xw + (long)dir * XW_DIR;
  const half_t* WF  = WhhF + (long)dir * KTR * NT * 512;
  int colBase = dir * H_;
  int lan15   = lane & 15;
  int khalfA  = (lane >> 4) * 8;
  int moff    = (lane >> 4) * 8;   // C-layout M offset
  const half_t* hrow = hS + (long)(wid * 16 + lan15) * KREC;

  for (int s = 0; s < T_; ++s) {
    int tt = dir ? (T_ - 1 - s) : s;
    // Load all A fragments (previous h) for this wave's batch rows.
    HV afr[KTR];
#pragma unroll
    for (int kt = 0; kt < KTR; ++kt) {
      afr[kt].h[0] = *(const v8h*)(hrow + kt * 32 + khalfA);
      afr[kt].h[1] = *(const v8h*)(hrow + kt * 32 + khalfA + 16);
    }
    const float* xwT = xwD + (long)tt * NT * 8 * 256 + (long)wid * 256 + lane * 8;
#pragma unroll 1
    for (int ht = 0; ht < 13; ++ht) {
      v8f acc[4];
#pragma unroll
      for (int g = 0; g < 4; ++g) {
        const float* p = xwT + (long)(g * 13 + ht) * 8 * 256;
        v4f lo = *(const v4f*)p, hi = *(const v4f*)(p + 4);
        acc[g][0] = lo[0]; acc[g][1] = lo[1]; acc[g][2] = lo[2]; acc[g][3] = lo[3];
        acc[g][4] = hi[0]; acc[g][5] = hi[1]; acc[g][6] = hi[2]; acc[g][7] = hi[3];
      }
#pragma unroll
      for (int kt = 0; kt < KTR; ++kt) {
#pragma unroll
        for (int g = 0; g < 4; ++g) {
          v16h b = *(const v16h*)(WF + ((long)kt * NT + (g * 13 + ht)) * 512 + lane * 16);
          acc[g] = __builtin_amdgcn_wmma_f32_16x16x32_f16(false, afr[kt].v, false, b,
                                                          (short)0, acc[g], false, false);
        }
      }
      float* cp = cS + ((long)(wid * 13 + ht) * 256) + lane * 8;
      int col = ht * 16 + lan15;
      int gm  = wid * 16 + moff;
#pragma unroll
      for (int r = 0; r < 8; ++r) {
        float iv = sigm(acc[0][r]);
        float fv = sigm(acc[1][r]);
        float gv = tanhf(acc[2][r]);
        float ov = sigm(acc[3][r]);
        float cn = fv * cp[r] + iv * gv;
        cp[r] = cn;
        float hv = ov * tanhf(cn);
        if (col < H_) {
          half_t h16 = (half_t)hv;
          hS[(long)(gm + r) * KREC + col] = h16;                         // feedback
          hOut[((long)tt * B_ + gm + r) * IN1 + colBase + col] = h16;    // layer output
        }
      }
    }
    __syncthreads();
  }
}

// ---------------------------------------------------------------------------
// Emission GEMM: em(T*B, 32) = h1(T*B, 416) * lin_w^T + lin_b, row-major f32.
// ---------------------------------------------------------------------------
__global__ __launch_bounds__(256) void wmma_gemm_em(const half_t* __restrict__ A,
                                                    const half_t* __restrict__ Bf,
                                                    const float* __restrict__ lin_b,
                                                    float* __restrict__ em) {
  int wid = threadIdx.x >> 5, lane = threadIdx.x & 31;
  long gw = (long)blockIdx.x * 8 + wid;
  if (gw >= (long)MT * 2) return;
  int nt = (int)(gw & 1);
  int mt = (int)(gw >> 1);
  int lan15 = lane & 15;
  int n = nt * 16 + lan15;
  float bv = (n < K_) ? lin_b[n] : 0.f;
  v8f acc = {bv, bv, bv, bv, bv, bv, bv, bv};
  int khalfA = (lane >> 4) * 8;
  const half_t* Arow = A + (long)(mt * 16 + lan15) * IN1;
#pragma unroll
  for (int kt = 0; kt < 13; ++kt) {
    HV a;
    a.h[0] = *(const v8h*)(Arow + kt * 32 + khalfA);
    a.h[1] = *(const v8h*)(Arow + kt * 32 + khalfA + 16);
    v16h b = *(const v16h*)(Bf + ((long)kt * 2 + nt) * 512 + lane * 16);
    acc = __builtin_amdgcn_wmma_f32_16x16x32_f16(false, a.v, false, b, (short)0, acc,
                                                 false, false);
  }
  int moff = (lane >> 4) * 8;
#pragma unroll
  for (int r = 0; r < 8; ++r)
    em[((long)mt * 16 + moff + r) * 32 + n] = acc[r];
}

// ---------------------------------------------------------------------------
// CRF log-likelihood: one wave per batch element; lane j = tag j (j < 25).
// Transition column cached in registers; cross-lane broadcast via shfl.
// ---------------------------------------------------------------------------
__global__ __launch_bounds__(256) void crf_kernel(const float* __restrict__ em,
                                                  const int* __restrict__ y,
                                                  const int* __restrict__ mask,
                                                  const float* __restrict__ start,
                                                  const float* __restrict__ endv,
                                                  const float* __restrict__ trans,
                                                  float* __restrict__ out) {
  int wid = threadIdx.x >> 5, lane = threadIdx.x & 31;
  int b = blockIdx.x * 8 + wid;
  if (b >= B_) return;
  int j = lane;
  float tc[K_];
#pragma unroll
  for (int i = 0; i < K_; ++i) tc[i] = (j < K_) ? trans[i * K_ + j] : 0.f;

  float s = (j < K_) ? (start[j] + em[(long)b * 32 + j]) : -1e30f;
  for (int t = 1; t < T_; ++t) {
    float m = -1e30f, sum = 0.f;
#pragma unroll
    for (int i = 0; i < K_; ++i) {
      float si = __shfl(s, i, 32);
      float v = si + tc[i];
      float nm = fmaxf(m, v);
      sum = sum * __expf(m - nm) + __expf(v - nm);
      m = nm;
    }
    float emt = em[((long)t * B_ + b) * 32 + ((j < K_) ? j : 0)];
    float ns = m + __logf(sum) + emt;
    if (j < K_ && mask[(long)b * T_ + t] != 0) s = ns;
  }
  // denominator: logsumexp over tags of s + end
  float v = (j < K_) ? (s + endv[j]) : -1e30f;
  float m = v;
  for (int off = 16; off; off >>= 1) m = fmaxf(m, __shfl_xor(m, off, 32));
  float e = __expf(v - m);
  for (int off = 16; off; off >>= 1) e += __shfl_xor(e, off, 32);
  float den = m + __logf(e);
  // numerator: lane-parallel over time
  float np = 0.f, cnt = 0.f;
  for (int t = lane; t < T_; t += 32) {
    float mk = (mask[(long)b * T_ + t] != 0) ? 1.f : 0.f;
    cnt += mk;
    if (t >= 1) {
      int yp = y[(long)b * T_ + t - 1];
      int yc = y[(long)b * T_ + t];
      np += mk * (trans[yp * K_ + yc] + em[((long)t * B_ + b) * 32 + yc]);
    }
  }
  for (int off = 16; off; off >>= 1) {
    np += __shfl_xor(np, off, 32);
    cnt += __shfl_xor(cnt, off, 32);
  }
  if (lane == 0) {
    int y0 = y[(long)b * T_];
    float num = np + start[y0] + em[(long)b * 32 + y0];
    int last = (int)cnt - 1;
    num += endv[y[(long)b * T_ + last]];
    atomicAdd(out, num - den);
  }
}

// ---------------------------------------------------------------------------
extern "C" void kernel_launch(void* const* d_in, const int* in_sizes, int n_in,
                              void* d_out, int out_size, void* d_ws, size_t ws_size,
                              hipStream_t stream) {
  const int*   x       = (const int*)d_in[0];
  const int*   yv      = (const int*)d_in[1];
  const int*   mask    = (const int*)d_in[2];
  const float* emb     = (const float*)d_in[3];
  const float* w_ih_l0 = (const float*)d_in[4];
  const float* w_hh_l0 = (const float*)d_in[5];
  const float* b_l0    = (const float*)d_in[6];
  const float* w_ih_l1 = (const float*)d_in[7];
  const float* w_hh_l1 = (const float*)d_in[8];
  const float* b_l1    = (const float*)d_in[9];
  const float* lin_w   = (const float*)d_in[10];
  const float* lin_b   = (const float*)d_in[11];
  const float* crf_s   = (const float*)d_in[12];
  const float* crf_e   = (const float*)d_in[13];
  const float* crf_t   = (const float*)d_in[14];
  float* out = (float*)d_out;

  // Carve workspace
  size_t off = 0;
  auto carve = [&](size_t bytes) -> char* {
    off = (off + 255) & ~(size_t)255;
    char* p = (char*)d_ws + off;
    off += bytes;
    return p;
  };
  half_t* xeh   = (half_t*)carve((size_t)T_ * B_ * Dp * 2);
  half_t* wih0f = (half_t*)carve((size_t)2 * 2 * NT * 512 * 2);
  half_t* whh0f = (half_t*)carve((size_t)2 * KTR * NT * 512 * 2);
  half_t* wih1f = (half_t*)carve((size_t)2 * 13 * NT * 512 * 2);
  half_t* whh1f = (half_t*)carve((size_t)2 * KTR * NT * 512 * 2);
  half_t* linf  = (half_t*)carve((size_t)13 * 2 * 512 * 2);
  float*  bias0 = (float*)carve((size_t)2 * 4 * Hp * 4);
  float*  bias1 = (float*)carve((size_t)2 * 4 * Hp * 4);
  float*  xwbuf = (float*)carve((size_t)2 * XW_DIR * 4);
  half_t* h0    = (half_t*)carve((size_t)T_ * B_ * IN1 * 2);
  half_t* h1    = (half_t*)carve((size_t)T_ * B_ * IN1 * 2);
  float*  embuf = (float*)carve((size_t)T_ * B_ * 32 * 4);

  constexpr size_t REC_LDS = (size_t)B_ * KREC * sizeof(half_t) + 8 * 13 * 256 * sizeof(float);
  (void)hipFuncSetAttribute((const void*)lstm_rec,
                            hipFuncAttributeMaxDynamicSharedMemorySize, (int)REC_LDS);

  zero_out<<<1, 64, 0, stream>>>(out);
  {
    long n = (long)T_ * B_ * IN1;
    zero_half<<<4096, 256, 0, stream>>>(h0, n);
    zero_half<<<4096, 256, 0, stream>>>(h1, n);
  }
  // Embedding
  {
    long total = (long)T_ * B_ * Dp;
    embed_kernel<<<(int)((total + 255) / 256), 256, 0, stream>>>(x, emb, xeh);
  }
  // Weight swizzles (B-fragment layouts) + biases
  {
    long t0 = (long)2 * 2 * NT * 512;
    prep_bfrag<<<(int)((t0 + 255) / 256), 256, 0, stream>>>(w_ih_l0, wih0f, 2, NT, H_, 13, D_, t0);
    long t1 = (long)2 * KTR * NT * 512;
    prep_bfrag<<<(int)((t1 + 255) / 256), 256, 0, stream>>>(w_hh_l0, whh0f, KTR, NT, H_, 13, H_, t1);
    long t2 = (long)2 * 13 * NT * 512;
    prep_bfrag<<<(int)((t2 + 255) / 256), 256, 0, stream>>>(w_ih_l1, wih1f, 13, NT, H_, 13, 2 * H_, t2);
    prep_bfrag<<<(int)((t1 + 255) / 256), 256, 0, stream>>>(w_hh_l1, whh1f, KTR, NT, H_, 13, H_, t1);
    long t3 = (long)13 * 2 * 512;
    prep_bfrag<<<(int)((t3 + 255) / 256), 256, 0, stream>>>(lin_w, linf, 13, 2, K_, 2, 2 * H_, t3);
    prep_bias<<<(2 * 4 * Hp + 255) / 256, 256, 0, stream>>>(b_l0, bias0);
    prep_bias<<<(2 * 4 * Hp + 255) / 256, 256, 0, stream>>>(b_l1, bias1);
  }
  // Layer 0: input projection + recurrence
  {
    long tw = (long)2 * MT * NT;
    wmma_gemm_xw<<<(int)((tw + 7) / 8), 256, 0, stream>>>(xeh, Dp, wih0f, 2, bias0, xwbuf, 2);
    lstm_rec<<<2, 256, REC_LDS, stream>>>(xwbuf, whh0f, h0);
  }
  // Layer 1: input projection + recurrence
  {
    long tw = (long)2 * MT * NT;
    wmma_gemm_xw<<<(int)((tw + 7) / 8), 256, 0, stream>>>(h0, IN1, wih1f, 13, bias1, xwbuf, 2);
    lstm_rec<<<2, 256, REC_LDS, stream>>>(xwbuf, whh1f, h1);
  }
  // Emissions
  {
    long tw = (long)MT * 2;
    wmma_gemm_em<<<(int)((tw + 7) / 8), 256, 0, stream>>>(h1, linf, lin_b, embuf);
  }
  // CRF
  crf_kernel<<<(B_ + 7) / 8, 256, 0, stream>>>(embuf, yv, mask, crf_s, crf_e, crf_t, out);
}
